// Attention_73950746903143
// MI455X (gfx1250) — matrix operand order repaired and checked by reading, not compiled, not run on hardware
//
#include <hip/hip_runtime.h>

// ---------------------------------------------------------------------------
// BEiT attention, MI455X (gfx1250): bf16 WMMA + async global->LDS staging,
// K-depth-64 double-buffered tiles, batched fragment loads.
// ---------------------------------------------------------------------------

typedef __bf16 bf16_t;
typedef __attribute__((ext_vector_type(4)))  bf16_t v4bf;
typedef __attribute__((ext_vector_type(8)))  bf16_t v8bf;
typedef __attribute__((ext_vector_type(16))) bf16_t v16bf;
typedef __attribute__((ext_vector_type(8)))  float  v8f;

#define NH 12
#define NT 1024
#define CC 768
#define DD 64

#define WAIT_ASYNC(n) asm volatile("s_wait_asynccnt " #n ::: "memory")

__device__ __forceinline__ v8f wmma_bf16(v16bf a, v16bf b, v8f c) {
  // (neg_a, A, neg_b, B, c_mod, C, reuse_a, reuse_b)
  return __builtin_amdgcn_wmma_f32_16x16x32_bf16(false, a, false, b, (short)0, c,
                                                 false, false);
}

// Async copy of 16 bytes global -> LDS (per lane). ASYNCcnt-tracked.
__device__ __forceinline__ void async_b128(unsigned lds_off, const void* gptr) {
  asm volatile("global_load_async_to_lds_b128 %0, %1, off"
               :: "v"(lds_off), "v"((unsigned long long)(uintptr_t)gptr)
               : "memory");
}
__device__ __forceinline__ unsigned lds_off_of(const void* p) {
  return (unsigned)(uintptr_t)p;   // LDS aperture: low 32 bits = LDS offset
}

// A-matrix 16x32 bf16 fragment (ISA 7.12.2): lane L -> row L&15,
// VGPR0-3 hold K = (L>>4)*8 .. +7, VGPR4-7 hold K = 16+(L>>4)*8 .. +7.
__device__ __forceinline__ v16bf frag_a(const bf16_t* base, int ld) {
  const int lane = threadIdx.x & 31;
  const int r = lane & 15, hh = lane >> 4;
  const bf16_t* p = base + r * ld + hh * 8;
  v8bf c0 = *(const v8bf*)(p);
  v8bf c1 = *(const v8bf*)(p + 16);
  return __builtin_shufflevector(c0, c1, 0, 1, 2, 3, 4, 5, 6, 7,
                                 8, 9, 10, 11, 12, 13, 14, 15);
}

// B-matrix 32x16 bf16 fragment: lane L -> column L&15, K = (L>>4)*8 .. +7, 16+(L>>4)*8 .. +7.
// Source row-major [n][k] (i.e. B = srcT): per lane two 8-element chunks.
__device__ __forceinline__ v16bf frag_b(const bf16_t* base, int ld) {
  const int lane = threadIdx.x & 31;
  const int r = lane & 15, hh = lane >> 4;
  const bf16_t* p = base + r * ld + hh * 8;
  v8bf c0 = *(const v8bf*)(p);
  v8bf c1 = *(const v8bf*)(p + 16);
  return __builtin_shufflevector(c0, c1, 0, 1, 2, 3, 4, 5, 6, 7,
                                 8, 9, 10, 11, 12, 13, 14, 15);
}

__device__ __forceinline__ v8f vzero8() {
  v8f z = {0.f, 0.f, 0.f, 0.f, 0.f, 0.f, 0.f, 0.f};
  return z;
}

// --------------------------- f32 -> bf16 convert ---------------------------
__global__ void cvt_f32_bf16_kernel(const float* __restrict__ in,
                                    bf16_t* __restrict__ out, int n4) {
  int i = blockIdx.x * blockDim.x + threadIdx.x;
  if (i >= n4) return;
  float4 f = ((const float4*)in)[i];
  v4bf o;
  o[0] = (bf16_t)f.x; o[1] = (bf16_t)f.y; o[2] = (bf16_t)f.z; o[3] = (bf16_t)f.w;
  ((v4bf*)out)[i] = o;
}

// ------------------- expand relative position bias table -------------------
// Bexp[h][n][m] = table[ rpi[n][m] * NH + h ]   (bf16)
__global__ void bias_expand_kernel(const long long* __restrict__ rpi,
                                   const float* __restrict__ table,
                                   bf16_t* __restrict__ bexp) {
  int i = blockIdx.x * blockDim.x + threadIdx.x;   // i in [0, NT*NT)
  long long idx = rpi[i];
  const float* t = table + (int)idx * NH;
#pragma unroll
  for (int h = 0; h < NH; ++h)
    bexp[(size_t)h * NT * NT + i] = (bf16_t)t[h];
}

// ---------------- shared helper: stage a 64x64 weight tile -----------------
// Tile = rows [n0, n0+64) x k [kc, kc+64) of row-major [Nout][CC] bf16 source,
// stored into LDS as [64][64] (ld=64).  128 threads x 4 async b128 = 8 KB.
__device__ __forceinline__ void stage_wtile(const bf16_t* __restrict__ w,
                                            int n0, int kc, bf16_t* buf) {
  const int t = threadIdx.x;                 // 0..127
#pragma unroll
  for (int i = 0; i < 4; ++i) {
    int c = t * 4 + i;                       // 512 chunks of 16 B
    int row = c >> 3, sub = c & 7;           // 8 chunks per 128 B row
    async_b128(lds_off_of(buf + row * 64 + sub * 8),
               w + (size_t)(n0 + row) * CC + kc + sub * 8);
  }
}

// -------- shared GEMM inner step: 16x64 wave tile, one 64-deep k chunk -----
__device__ __forceinline__ void gemm_step(const bf16_t* __restrict__ a_src,
                                          const bf16_t* __restrict__ bb,
                                          v8f c[4]) {
  v16bf a0 = frag_a(a_src, CC);
  v16bf a1 = frag_a(a_src + 32, CC);
  v16bf bfg[8];
#pragma unroll
  for (int t = 0; t < 4; ++t) {
    bfg[2 * t]     = frag_b(bb + (16 * t) * 64, 64);
    bfg[2 * t + 1] = frag_b(bb + (16 * t) * 64 + 32, 64);
  }
#pragma unroll
  for (int t = 0; t < 4; ++t) {
    c[t] = wmma_bf16(a0, bfg[2 * t], c[t]);
    c[t] = wmma_bf16(a1, bfg[2 * t + 1], c[t]);
  }
}

// ------------------------------- QKV GEMM ----------------------------------
// qkv[m][j] = sum_k x[m][k] * W[j][k] + bias(j);  M=8192, Nout=2304, K=768.
__global__ __launch_bounds__(128) void qkv_gemm_kernel(
    const bf16_t* __restrict__ xb, const bf16_t* __restrict__ wb,
    const float* __restrict__ qbias, const float* __restrict__ vbias,
    bf16_t* __restrict__ Qb, bf16_t* __restrict__ Kb, bf16_t* __restrict__ Vt) {
  __shared__ __align__(16) bf16_t bbuf[2][64 * 64];
  const int wave = threadIdx.x >> 5;
  const int lane = threadIdx.x & 31;
  const int r = lane & 15, hh = lane >> 4;
  const int m0 = blockIdx.x * 64 + wave * 16;
  const int n0 = blockIdx.y * 64;

  v8f c[4];
#pragma unroll
  for (int t = 0; t < 4; ++t) c[t] = vzero8();

  stage_wtile(wb, n0, 0, bbuf[0]);
  int cur = 0;
#pragma unroll 1
  for (int kc = 0; kc < CC; kc += 64) {
    __syncthreads();                          // readers of buf[cur^1] done
    if (kc + 64 < CC) {
      stage_wtile(wb, n0, kc + 64, bbuf[cur ^ 1]);
      WAIT_ASYNC(4);                          // current tile complete
    } else {
      WAIT_ASYNC(0);
    }
    __syncthreads();                          // tile visible to all waves
    gemm_step(xb + (size_t)m0 * CC + kc, bbuf[cur], c);
    cur ^= 1;
  }

#pragma unroll
  for (int t = 0; t < 4; ++t) {
    int j = n0 + 16 * t + r;        // output feature in [0, 2304)
    int seg = j / CC;               // 0:Q 1:K 2:V
    int jj = j - seg * CC;
    int head = jj >> 6, dc = jj & 63;
    float bias = (seg == 0) ? qbias[jj] : ((seg == 2) ? vbias[jj] : 0.f);
#pragma unroll
    for (int v = 0; v < 8; ++v) {
      int m = m0 + hh * 8 + v;      // global row = b*NT + n
      int b = m >> 10, n = m & (NT - 1);
      float val = c[t][v] + bias;
      size_t bhn = ((size_t)(b * NH + head) * NT + n) * DD + dc;
      if (seg == 0) {
        Qb[bhn] = (bf16_t)(val * 0.125f);            // d^-0.5 = 1/8
      } else if (seg == 1) {
        Kb[bhn] = (bf16_t)val;
      } else {
        Vt[((size_t)(b * NH + head) * DD + dc) * NT + n] = (bf16_t)val;
      }
    }
  }
}

// --------------------------- flash attention core --------------------------
// grid = (NT/64, B*H), block = 128 (4 waves); each wave owns 16 query rows.
// 64-key tiles: K tile [64 tok][64 d], V^T tile [64 d][64 key], both 8 KB,
// double-buffered via async loads; P (16x64) reshaped through LDS (ldp=72).
__global__ __launch_bounds__(128) void flash_attn_kernel(
    const bf16_t* __restrict__ Qb, const bf16_t* __restrict__ Kb,
    const bf16_t* __restrict__ Vt, const bf16_t* __restrict__ bexp,
    bf16_t* __restrict__ outb) {
  __shared__ __align__(16) bf16_t kbuf[2][64 * 64];
  __shared__ __align__(16) bf16_t vbuf[2][64 * 64];
  __shared__ __align__(16) bf16_t plds[4][16 * 72];
  const int wave = threadIdx.x >> 5;
  const int lane = threadIdx.x & 31;
  const int tid  = threadIdx.x;
  const int r = lane & 15, hh = lane >> 4;
  const int bh = blockIdx.y;                 // b*NH + h
  const int h = bh % NH;
  const int b = bh / NH;
  const int q0 = blockIdx.x * 64 + wave * 16;

  const bf16_t* Qh = Qb + (size_t)bh * NT * DD;
  const bf16_t* Kh = Kb + (size_t)bh * NT * DD;
  const bf16_t* Vh = Vt + (size_t)bh * DD * NT;
  const bf16_t* Bs = bexp + (size_t)h * NT * NT;

  // stage 64-key K/V tile (8 async b128 per thread = 16 KB per block)
  auto stage_kv = [&](int kt, bf16_t* kb, bf16_t* vb) {
    const bf16_t* gk = Kh + (size_t)kt * DD;   // 8 KB contiguous
#pragma unroll
    for (int i = 0; i < 4; ++i) {
      int c = tid * 4 + i;
      async_b128(lds_off_of(kb + c * 8), gk + c * 8);
    }
#pragma unroll
    for (int i = 0; i < 4; ++i) {
      int c = tid * 4 + i;                     // 512 chunks of 16 B
      int row = c >> 3, sub = c & 7;           // 8 chunks per 128 B row
      async_b128(lds_off_of(vb + row * 64 + sub * 8),
                 Vh + (size_t)row * NT + kt + sub * 8);
    }
  };

  // Q fragments for the two 32-wide d chunks (held for the whole loop)
  v16bf aq0 = frag_a(Qh + (size_t)q0 * DD + 0, DD);
  v16bf aq1 = frag_a(Qh + (size_t)q0 * DD + 32, DD);

  v8f o[4];
#pragma unroll
  for (int t = 0; t < 4; ++t) o[t] = vzero8();
  float mrun[8], lrun[8];
#pragma unroll
  for (int v = 0; v < 8; ++v) { mrun[v] = -1e30f; lrun[v] = 0.f; }

  bf16_t* myp = plds[wave];
  stage_kv(0, kbuf[0], vbuf[0]);
  int cur = 0;

#pragma unroll 1
  for (int kt = 0; kt < NT; kt += 64) {
    __syncthreads();                          // readers of buf[cur^1] done
    if (kt + 64 < NT) {
      stage_kv(kt + 64, kbuf[cur ^ 1], vbuf[cur ^ 1]);
      WAIT_ASYNC(8);                          // current tile complete
    } else {
      WAIT_ASYNC(0);
    }
    __syncthreads();                          // tile visible to all waves
    const bf16_t* kc_ = kbuf[cur];
    const bf16_t* vc_ = vbuf[cur];

    // S = (Q * d^-0.5) . K^T for 16x64 key tile, f32 accum (batched frags)
    v8f s[4];
    {
      v16bf bk[8];
#pragma unroll
      for (int ct = 0; ct < 4; ++ct) {
        bk[2 * ct]     = frag_b(kc_ + (16 * ct) * 64, 64);
        bk[2 * ct + 1] = frag_b(kc_ + (16 * ct) * 64 + 32, 64);
      }
#pragma unroll
      for (int ct = 0; ct < 4; ++ct) {
        s[ct] = wmma_bf16(aq0, bk[2 * ct], vzero8());
        s[ct] = wmma_bf16(aq1, bk[2 * ct + 1], s[ct]);
      }
    }

    // + relative position bias (C-layout: row = q0+hh*8+v, col = kt+16ct+r)
#pragma unroll
    for (int ct = 0; ct < 4; ++ct)
#pragma unroll
      for (int v = 0; v < 8; ++v) {
        int qr = q0 + hh * 8 + v;
        s[ct][v] += (float)Bs[(size_t)qr * NT + kt + 16 * ct + r];
      }

    // online softmax: stats per row, reduced across the 16-lane half
#pragma unroll
    for (int v = 0; v < 8; ++v) {
      float t = fmaxf(fmaxf(s[0][v], s[1][v]), fmaxf(s[2][v], s[3][v]));
      t = fmaxf(t, __shfl_xor(t, 1, 32));
      t = fmaxf(t, __shfl_xor(t, 2, 32));
      t = fmaxf(t, __shfl_xor(t, 4, 32));
      t = fmaxf(t, __shfl_xor(t, 8, 32));
      float mn = fmaxf(mrun[v], t);
      float sc = __expf(mrun[v] - mn);
      lrun[v] *= sc;
      o[0][v] *= sc; o[1][v] *= sc; o[2][v] *= sc; o[3][v] *= sc;
      float e0 = __expf(s[0][v] - mn);
      float e1 = __expf(s[1][v] - mn);
      float e2 = __expf(s[2][v] - mn);
      float e3 = __expf(s[3][v] - mn);
      s[0][v] = e0; s[1][v] = e1; s[2][v] = e2; s[3][v] = e3;
      float ls = (e0 + e1) + (e2 + e3);
      ls += __shfl_xor(ls, 1, 32);
      ls += __shfl_xor(ls, 2, 32);
      ls += __shfl_xor(ls, 4, 32);
      ls += __shfl_xor(ls, 8, 32);
      lrun[v] += ls;
      mrun[v] = mn;
    }

    // C-layout -> A-layout reshape of P (16x64) through LDS (bf16, ldp=72)
    __syncthreads();
#pragma unroll
    for (int ct = 0; ct < 4; ++ct)
#pragma unroll
      for (int v = 0; v < 8; ++v)
        myp[(hh * 8 + v) * 72 + 16 * ct + r] = (bf16_t)s[ct][v];
    __syncthreads();
    v16bf pa0 = frag_a(myp, 72);
    v16bf pa1 = frag_a(myp + 32, 72);

    // O += P . V  (B fragments from V^T tile in LDS, ld=64; batched)
    {
      v16bf bv[8];
#pragma unroll
      for (int dt = 0; dt < 4; ++dt) {
        bv[2 * dt]     = frag_b(vc_ + (16 * dt) * 64, 64);
        bv[2 * dt + 1] = frag_b(vc_ + (16 * dt) * 64 + 32, 64);
      }
#pragma unroll
      for (int dt = 0; dt < 4; ++dt) {
        o[dt] = wmma_bf16(pa0, bv[2 * dt], o[dt]);
        o[dt] = wmma_bf16(pa1, bv[2 * dt + 1], o[dt]);
      }
    }

    cur ^= 1;
  }

  // finalize: divide by softmax denom, emit bf16 [B*N, C] for proj GEMM
#pragma unroll
  for (int v = 0; v < 8; ++v) {
    float inv = 1.f / lrun[v];
    o[0][v] *= inv; o[1][v] *= inv; o[2][v] *= inv; o[3][v] *= inv;
  }
#pragma unroll
  for (int dt = 0; dt < 4; ++dt) {
#pragma unroll
    for (int v = 0; v < 8; ++v) {
      int tok = q0 + hh * 8 + v;
      outb[((size_t)b * NT + tok) * CC + h * DD + dt * 16 + r] = (bf16_t)o[dt][v];
    }
  }
}

// ------------------------------- proj GEMM ---------------------------------
__global__ __launch_bounds__(128) void proj_gemm_kernel(
    const bf16_t* __restrict__ ab, const bf16_t* __restrict__ wb,
    const float* __restrict__ bias, float* __restrict__ out) {
  __shared__ __align__(16) bf16_t bbuf[2][64 * 64];
  const int wave = threadIdx.x >> 5;
  const int lane = threadIdx.x & 31;
  const int r = lane & 15, hh = lane >> 4;
  const int m0 = blockIdx.x * 64 + wave * 16;
  const int n0 = blockIdx.y * 64;

  v8f c[4];
#pragma unroll
  for (int t = 0; t < 4; ++t) c[t] = vzero8();

  stage_wtile(wb, n0, 0, bbuf[0]);
  int cur = 0;
#pragma unroll 1
  for (int kc = 0; kc < CC; kc += 64) {
    __syncthreads();
    if (kc + 64 < CC) {
      stage_wtile(wb, n0, kc + 64, bbuf[cur ^ 1]);
      WAIT_ASYNC(4);
    } else {
      WAIT_ASYNC(0);
    }
    __syncthreads();
    gemm_step(ab + (size_t)m0 * CC + kc, bbuf[cur], c);
    cur ^= 1;
  }

#pragma unroll
  for (int t = 0; t < 4; ++t) {
    int j = n0 + 16 * t + r;
    float bv = bias[j];
#pragma unroll
    for (int v = 0; v < 8; ++v) {
      int m = m0 + hh * 8 + v;
      out[(size_t)m * CC + j] = c[t][v] + bv;
    }
  }
}

// ---------------------------------------------------------------------------
extern "C" void kernel_launch(void* const* d_in, const int* in_sizes, int n_in,
                              void* d_out, int out_size, void* d_ws, size_t ws_size,
                              hipStream_t stream) {
  const float*     x     = (const float*)d_in[0];
  const long long* rpi   = (const long long*)d_in[1];   // int64 indices
  const float*     wqkv  = (const float*)d_in[2];
  const float*     qbias = (const float*)d_in[3];
  const float*     vbias = (const float*)d_in[4];
  const float*     wproj = (const float*)d_in[5];
  const float*     pbias = (const float*)d_in[6];
  const float*     table = (const float*)d_in[7];
  float* out = (float*)d_out;

  size_t off = 0;
  char* ws = (char*)d_ws;
  auto wsalloc = [&](size_t bytes) -> void* {
    void* p = ws + off;
    off += (bytes + 255) & ~(size_t)255;
    return p;
  };
  const size_t M = 8ull * NT;                       // 8192
  bf16_t* xb   = (bf16_t*)wsalloc(M * CC * 2);
  bf16_t* wqb  = (bf16_t*)wsalloc(3ull * CC * CC * 2);
  bf16_t* wpb  = (bf16_t*)wsalloc((size_t)CC * CC * 2);
  bf16_t* Qb   = (bf16_t*)wsalloc(M * CC * 2);      // [B,H,N,d]
  bf16_t* Kb   = (bf16_t*)wsalloc(M * CC * 2);      // [B,H,N,d]
  bf16_t* Vt   = (bf16_t*)wsalloc(M * CC * 2);      // [B,H,d,N]
  bf16_t* Bexp = (bf16_t*)wsalloc((size_t)NH * NT * NT * 2);
  bf16_t* Ab   = (bf16_t*)wsalloc(M * CC * 2);
  (void)ws_size; (void)in_sizes; (void)n_in; (void)out_size;

  // 1) f32 -> bf16 converts
  cvt_f32_bf16_kernel<<<(int)(M * CC / 4 / 256), 256, 0, stream>>>(x, xb, (int)(M * CC / 4));
  cvt_f32_bf16_kernel<<<3 * CC * CC / 4 / 256, 256, 0, stream>>>(wqkv, wqb, 3 * CC * CC / 4);
  cvt_f32_bf16_kernel<<<CC * CC / 4 / 256, 256, 0, stream>>>(wproj, wpb, CC * CC / 4);

  // 2) relative position bias expansion -> [H, N, N] bf16 (L2-resident, 25 MB)
  bias_expand_kernel<<<NT * NT / 256, 256, 0, stream>>>(rpi, table, Bexp);

  // 3) QKV GEMM (M=8192, Nout=2304, K=768)
  qkv_gemm_kernel<<<dim3((int)(M / 64), 3 * CC / 64), 128, 0, stream>>>(
      xb, wqb, qbias, vbias, Qb, Kb, Vt);

  // 4) flash attention (per (b,h), 16 query rows per wave)
  flash_attn_kernel<<<dim3(NT / 64, 8 * NH), 128, 0, stream>>>(Qb, Kb, Vt, Bexp, Ab);

  // 5) projection GEMM (M=8192, Nout=768, K=768) + bias -> f32 out
  proj_gemm_kernel<<<dim3((int)(M / 64), CC / 64), 128, 0, stream>>>(Ab, wpb, pbias, out);
}
